// kNN_42863773614091
// MI455X (gfx1250) — matrix operand order repaired and checked by reading, not compiled, not run on hardware
//
#include <hip/hip_runtime.h>
#include <hip/hip_fp16.h>

typedef __attribute__((ext_vector_type(16))) _Float16 v16h;
typedef __attribute__((ext_vector_type(8)))  _Float16 v8h;
typedef __attribute__((ext_vector_type(8)))  float    v8f;

#define B_ROWS 2048
#define N_SUP  50000
#define DIM    1024
#define BM 128
#define BN 128
#define BK 32
#define LDK 40   // padded halfs per LDS row (80B stride -> conflict-free)

#if __has_builtin(__builtin_amdgcn_tensor_load_to_lds)
#define USE_TDM 1
#else
#define USE_TDM 0
#endif

// ---------- helpers ----------

// order-preserving float -> uint mapping (min of keys == min of floats)
__device__ __forceinline__ unsigned int fkey(float f) {
    unsigned int u = __float_as_uint(f);
    return (u & 0x80000000u) ? ~u : (u | 0x80000000u);
}

__device__ __forceinline__ v16h combine16(v8h lo, v8h hi) {
    v16h r;
#pragma unroll
    for (int t = 0; t < 8; ++t) { r[t] = lo[t]; r[8 + t] = hi[t]; }
    return r;
}

#if USE_TDM
typedef __attribute__((ext_vector_type(4))) unsigned int u32x4;
typedef __attribute__((ext_vector_type(8))) int i32x8;
typedef __attribute__((ext_vector_type(4))) int i32x4;

// LDS byte offset (0-based within workgroup LDS block) of a shared pointer
__device__ __forceinline__ unsigned int lds_addr_of(const void* p) {
    return (unsigned int)(unsigned long long)
           (__attribute__((address_space(3))) const void*)p;
}

// 2D TDM load: tileRows x BK f16 tile from a row-major [tensorRows x DIM]
// tensor into LDS, hardware-padded to LDK(=40)-half rows.
// pad_interval=3 -> pad every 16 DWORDs (64B = one 32-half row)
// pad_amount=3   -> insert 4 DWORDs (16B = 8 halfs) of padding
__device__ __forceinline__ void tdm_load_2d(const _Float16* gtile,
                                            unsigned int lds_byte_addr,
                                            int tensorRows, int tileRows) {
    unsigned long long ga = (unsigned long long)(const void*)gtile;
    u32x4 g0;
    g0[0] = 1u;                                               // count=1 (valid user D#)
    g0[1] = lds_byte_addr;                                    // lds_addr[31:0]
    g0[2] = (unsigned int)(ga & 0xFFFFFFFFull);               // global_addr[31:0]
    g0[3] = (unsigned int)((ga >> 32) & 0x01FFFFFFull)        // global_addr[56:32]
            | (2u << 30);                                     // type=2 ("image")
    i32x8 g1;
    g1[0] = (1 << 16)      // data_size = 1 -> 2 bytes
          | (1 << 20)      // pad_enable
          | (3 << 22)      // pad_interval: 16 DWORDs
          | (3 << 25);     // pad_amount: 4 DWORDs
    g1[1] = (DIM & 0xFFFF) << 16;                             // tensor_dim0[15:0]
    g1[2] = ((DIM >> 16) & 0xFFFF)                            // tensor_dim0[31:16]
          | ((tensorRows & 0xFFFF) << 16);                    // tensor_dim1[15:0]
    g1[3] = ((tensorRows >> 16) & 0xFFFF)                     // tensor_dim1[31:16]
          | (BK << 16);                                       // tile_dim0 = 32
    g1[4] = tileRows & 0xFFFF;                                // tile_dim1 (tile_dim2=0)
    g1[5] = DIM;                                              // tensor_dim0_stride[31:0]
    g1[6] = 0;
    g1[7] = 0;
    i32x4 gz;
    gz[0] = 0; gz[1] = 0; gz[2] = 0; gz[3] = 0;
    i32x8 gz8;
#pragma unroll
    for (int t = 0; t < 8; ++t) gz8[t] = 0;
    // 6-arg form (clang-23 / therock-10 headers): groups 2/3 unused for 2D
    // tiles (tile_dim2 == 0), trailing group + cpol zeroed.
    __builtin_amdgcn_tensor_load_to_lds(g0, g1, gz, gz, gz8, 0);
}
#endif

// ---------- kernel 0: init per-row (key,idx) minima ----------
__global__ void init_rowmin(unsigned long long* rowmin) {
    int i = blockIdx.x * blockDim.x + threadIdx.x;
    if (i < B_ROWS) rowmin[i] = ~0ull;
}

// ---------- kernel 1: convert x fp32 -> f16 ----------
__global__ void cvt_x(const float* __restrict__ x, _Float16* __restrict__ xh) {
    int i = blockIdx.x * blockDim.x + threadIdx.x;
    xh[i] = (_Float16)x[i];
}

// ---------- kernel 2: convert support fp32 -> f16 and compute row norms ----------
__global__ __launch_bounds__(256) void cvt_s(const float* __restrict__ s,
                                             _Float16* __restrict__ sh,
                                             float* __restrict__ s2) {
    const int row = blockIdx.x;
    const int t = threadIdx.x;
    const float* rp = s + (size_t)row * DIM;
    _Float16* wp = sh + (size_t)row * DIM;
    float acc = 0.0f;
#pragma unroll
    for (int q = 0; q < DIM / 256; ++q) {
        float v = rp[t + q * 256];
        wp[t + q * 256] = (_Float16)v;
        acc += v * v;
    }
#pragma unroll
    for (int off = 16; off > 0; off >>= 1) acc += __shfl_xor(acc, off, 32);
    __shared__ float part[8];
    if ((t & 31) == 0) part[t >> 5] = acc;
    __syncthreads();
    if (t == 0) {
        float sum = 0.0f;
#pragma unroll
        for (int w = 0; w < 8; ++w) sum += part[w];
        s2[row] = sum;
    }
}

// ---------- kernel 3: tiled WMMA GEMM + per-row argmin reduction ----------
__global__ __launch_bounds__(256) void knn_gemm(const _Float16* __restrict__ xh,
                                                const _Float16* __restrict__ sh,
                                                const float* __restrict__ s2,
                                                unsigned long long* __restrict__ rowmin) {
    __shared__ _Float16 ldsA[2 * BM * LDK];   // double-buffered
    __shared__ _Float16 ldsB[2 * BN * LDK];
    __shared__ unsigned long long lmin[BM];

    const int tid  = threadIdx.x;
    const int wave = tid >> 5;          // 8 waves: 4 (M) x 2 (N)
    const int lane = tid & 31;
    const int waveM = wave >> 1;
    const int waveN = wave & 1;
    const int m16  = lane & 15;
    const int half = lane >> 4;
    const int blockRow = blockIdx.y * BM;
    const int blockCol = blockIdx.x * BN;

    if (tid < BM) lmin[tid] = ~0ull;

    v8f acc[2][4];
#pragma unroll
    for (int i = 0; i < 2; ++i)
#pragma unroll
        for (int j = 0; j < 4; ++j)
#pragma unroll
            for (int t = 0; t < 8; ++t) acc[i][j][t] = 0.0f;

#if USE_TDM
    // ===================== TDM (async tensor DMA) path =====================
    const int validB = (N_SUP - blockCol < BN) ? (N_SUP - blockCol) : BN;

    // zero-fill LDS rows TDM will never write (tail block) in both buffers
    if (validB < BN) {
        v8h z;
#pragma unroll
        for (int t = 0; t < 8; ++t) z[t] = (_Float16)0.0f;
        for (int r = validB + tid; r < BN; r += 256) {
            _Float16* p0 = ldsB + r * LDK;
            _Float16* p1 = ldsB + BN * LDK + r * LDK;
#pragma unroll
            for (int sgm = 0; sgm < 5; ++sgm) {
                *(v8h*)(p0 + 8 * sgm) = z;
                *(v8h*)(p1 + 8 * sgm) = z;
            }
        }
    }

    const _Float16* gA = xh + (size_t)blockRow * DIM;
    const _Float16* gB = sh + (size_t)blockCol * DIM;
    const unsigned int ldsA_addr = lds_addr_of(ldsA);
    const unsigned int ldsB_addr = lds_addr_of(ldsB);
    const unsigned int bufA_bytes = BM * LDK * sizeof(_Float16);
    const unsigned int bufB_bytes = BN * LDK * sizeof(_Float16);

    if (wave == 0) {                        // prologue: fill buffer 0
        tdm_load_2d(gA, ldsA_addr, B_ROWS, BM);
        tdm_load_2d(gB, ldsB_addr, N_SUP, validB);
    }

    for (int k0 = 0; k0 < DIM; k0 += BK) {
        const int cur = (k0 >> 5) & 1;
        __syncthreads();                    // all waves done reading buf[1-cur]
        if (wave == 0) {
            if (k0 + BK < DIM) {            // prefetch next pair into buf[1-cur]
                const int nxt = cur ^ 1;
                tdm_load_2d(gA + k0 + BK, ldsA_addr + (unsigned)nxt * bufA_bytes,
                            B_ROWS, BM);
                tdm_load_2d(gB + k0 + BK, ldsB_addr + (unsigned)nxt * bufB_bytes,
                            N_SUP, validB);
                __builtin_amdgcn_s_wait_tensorcnt(2);   // current pair complete
            } else {
                __builtin_amdgcn_s_wait_tensorcnt(0);
            }
        }
        __syncthreads();                    // buf[cur] published to all waves

        const _Float16* bA = ldsA + cur * BM * LDK;
        const _Float16* bB = ldsB + cur * BN * LDK;

        v16h afrag[2], bfrag[4];
#pragma unroll
        for (int i = 0; i < 2; ++i) {
            const _Float16* base = bA + (waveM * 32 + i * 16 + m16) * LDK + half * 8;
            afrag[i] = combine16(*(const v8h*)base, *(const v8h*)(base + 16));
        }
#pragma unroll
        for (int j = 0; j < 4; ++j) {
            const _Float16* base = bB + (waveN * 64 + j * 16 + m16) * LDK + half * 8;
            bfrag[j] = combine16(*(const v8h*)base, *(const v8h*)(base + 16));
        }
#pragma unroll
        for (int i = 0; i < 2; ++i)
#pragma unroll
            for (int j = 0; j < 4; ++j)
                acc[i][j] = __builtin_amdgcn_wmma_f32_16x16x32_f16(
                    false, afrag[i], false, bfrag[j], (short)0, acc[i][j], false, false);
    }
#else
    // ===================== fallback: per-lane staging path =====================
    const int ldRow = tid >> 1;          // 0..127
    const int ldSeg = (tid & 1) * 16;    // 0 or 16 halfs (32B)

    for (int k0 = 0; k0 < DIM; k0 += BK) {
        __syncthreads();
        {
            const _Float16* g = xh + (size_t)(blockRow + ldRow) * DIM + k0 + ldSeg;
            v8h a0 = *(const v8h*)g;
            v8h a1 = *(const v8h*)(g + 8);
            _Float16* l = ldsA + ldRow * LDK + ldSeg;
            *(v8h*)l       = a0;
            *(v8h*)(l + 8) = a1;
        }
        {
            const int gn = blockCol + ldRow;
            v8h b0, b1;
#pragma unroll
            for (int t = 0; t < 8; ++t) { b0[t] = (_Float16)0.0f; b1[t] = (_Float16)0.0f; }
            if (gn < N_SUP) {
                const _Float16* g = sh + (size_t)gn * DIM + k0 + ldSeg;
                b0 = *(const v8h*)g;
                b1 = *(const v8h*)(g + 8);
            }
            _Float16* l = ldsB + ldRow * LDK + ldSeg;
            *(v8h*)l       = b0;
            *(v8h*)(l + 8) = b1;
        }
        __syncthreads();

        v16h afrag[2], bfrag[4];
#pragma unroll
        for (int i = 0; i < 2; ++i) {
            const _Float16* base = ldsA + (waveM * 32 + i * 16 + m16) * LDK + half * 8;
            afrag[i] = combine16(*(const v8h*)base, *(const v8h*)(base + 16));
        }
#pragma unroll
        for (int j = 0; j < 4; ++j) {
            const _Float16* base = ldsB + (waveN * 64 + j * 16 + m16) * LDK + half * 8;
            bfrag[j] = combine16(*(const v8h*)base, *(const v8h*)(base + 16));
        }
#pragma unroll
        for (int i = 0; i < 2; ++i)
#pragma unroll
            for (int j = 0; j < 4; ++j)
                acc[i][j] = __builtin_amdgcn_wmma_f32_16x16x32_f16(
                    false, afrag[i], false, bfrag[j], (short)0, acc[i][j], false, false);
    }
#endif

    // ---- score = s2[n] - 2*dot ; per-lane -> LDS per-row argmin ----
    float s2v[4];
    int gncol[4];
#pragma unroll
    for (int j = 0; j < 4; ++j) {
        int gn = blockCol + waveN * 64 + j * 16 + m16;
        gncol[j] = gn;
        s2v[j] = (gn < N_SUP) ? s2[gn] : __builtin_inff();
    }
#pragma unroll
    for (int i = 0; i < 2; ++i) {
#pragma unroll
        for (int r = 0; r < 8; ++r) {
            const int rloc = waveM * 32 + i * 16 + half * 8 + r;
            unsigned long long best = ~0ull;
#pragma unroll
            for (int j = 0; j < 4; ++j) {
                float sc = s2v[j] - 2.0f * acc[i][j][r];
                unsigned long long p =
                    ((unsigned long long)fkey(sc) << 32) | (unsigned int)gncol[j];
                best = (p < best) ? p : best;
            }
            atomicMin(&lmin[rloc], best);
        }
    }
    __syncthreads();
    if (tid < BM) atomicMin(&rowmin[blockRow + tid], lmin[tid]);
}

// ---------- kernel 4: gather label, emit one-hot ----------
__global__ void finalize(const unsigned long long* __restrict__ rowmin,
                         const int* __restrict__ labels, float* __restrict__ out) {
    int b = blockIdx.x * blockDim.x + threadIdx.x;
    if (b < B_ROWS) {
        unsigned int idx = (unsigned int)(rowmin[b] & 0xFFFFFFFFu);
        if (idx >= N_SUP) idx = 0;
        int lab = labels[idx];
        out[b * 2 + 0] = (lab == 0) ? 1.0f : 0.0f;
        out[b * 2 + 1] = (lab == 1) ? 1.0f : 0.0f;
    }
}

extern "C" void kernel_launch(void* const* d_in, const int* in_sizes, int n_in,
                              void* d_out, int out_size, void* d_ws, size_t ws_size,
                              hipStream_t stream) {
    const float* x      = (const float*)d_in[0];
    const float* sup    = (const float*)d_in[1];
    const int*   labels = (const int*)d_in[2];
    float* out = (float*)d_out;

    char* ws = (char*)d_ws;
    const size_t XH_BYTES = (size_t)B_ROWS * DIM * sizeof(_Float16);   //   4 MB
    const size_t SH_BYTES = (size_t)N_SUP  * DIM * sizeof(_Float16);   // 100 MB (L2-resident)
    const size_t S2_BYTES = (size_t)N_SUP * sizeof(float);             // 200 KB
    _Float16* xh  = (_Float16*)ws;
    _Float16* shh = (_Float16*)(ws + XH_BYTES);
    float*    s2  = (float*)(ws + XH_BYTES + SH_BYTES);
    unsigned long long* rowmin =
        (unsigned long long*)(ws + XH_BYTES + SH_BYTES + S2_BYTES);

    init_rowmin<<<(B_ROWS + 255) / 256, 256, 0, stream>>>(rowmin);
    cvt_x<<<(B_ROWS * DIM) / 256, 256, 0, stream>>>(x, xh);
    cvt_s<<<N_SUP, 256, 0, stream>>>(sup, shh, s2);

    dim3 grid((N_SUP + BN - 1) / BN, B_ROWS / BM);   // 391 x 16
    knn_gemm<<<grid, 256, 0, stream>>>(xh, shh, s2, rowmin);

    finalize<<<(B_ROWS + 255) / 256, 256, 0, stream>>>(rowmin, labels, out);
}
